// BiLSTM_80401787781853
// MI455X (gfx1250) — matrix operand order repaired and checked by reading, not compiled, not run on hardware
//
#include <hip/hip_runtime.h>

// ---------------------------------------------------------------------------
// BiLSTM for MI455X (gfx1250, wave32, WMMA 16x16x32 f16, TDM gate streaming)
//   D=1024, H=512, B=16, T=2048, 2 layers x 2 directions
// ---------------------------------------------------------------------------

#define D_MODEL 1024
#define HIDDEN  512
#define GATES   2048      // 4*HIDDEN
#define BATCH   16
#define SEQ     2048
#define NTILES  128       // GATES/16

#define GSTEP_BYTES  131072u              // one timestep of gates: 128*32*8*4
#define HBUF_BYTES   16384u               // one h buffer: 16 x 512 fp16
#define GBUF_OFF     (2u * HBUF_BYTES)    // gate buffers start after h0,h1
#define SCAN_LDS     (GBUF_OFF + 2u * GSTEP_BYTES)   // 288 KB total

typedef _Float16 v16h __attribute__((ext_vector_type(16)));
typedef float    v8f  __attribute__((ext_vector_type(8)));
typedef unsigned int u32x4 __attribute__((ext_vector_type(4)));
typedef int          i32x4 __attribute__((ext_vector_type(4)));
typedef int          i32x8 __attribute__((ext_vector_type(8)));

union H16 { v16h v; uint4 u[2]; };
union F8  { v8f  f; uint4 u[2]; };

#if __has_builtin(__builtin_amdgcn_tensor_load_to_lds) && \
    __has_builtin(__builtin_amdgcn_s_wait_tensorcnt)
#define USE_TDM 1
#else
#define USE_TDM 0
#endif

__device__ __forceinline__ v8f wmma_f16(v16h a, v16h b, v8f c) {
  return __builtin_amdgcn_wmma_f32_16x16x32_f16(
      /*neg_a=*/false, a, /*neg_b=*/false, b,
      /*c_mod=*/(short)0, c, /*reuse_a=*/false, /*reuse_b=*/false);
}

__device__ __forceinline__ float sigmoidf_(float x) {
  return 1.0f / (1.0f + __expf(-x));
}
__device__ __forceinline__ float tanhf_(float x) {
  float t = __expf(2.0f * x);
  return 1.0f - 2.0f / (t + 1.0f);   // saturates correctly for |x| large
}

#if USE_TDM
// ---------------------------------------------------------------------------
// TDM: 1D DMA of one timestep's gate block (131072 B = 16384 x 8B elements)
// from global into LDS at byte offset `lds_off`. Issued once per wave.
// D# per CDNA5 ISA 8.3/8.4: group0 {count=1, lds_addr, global_addr, type=2},
// group1 {data_size=3 (8B), tensor_dim0=tile_dim0=16384, dim0_stride=16384}.
// ---------------------------------------------------------------------------
__device__ __forceinline__ void tdm_load_gates(const float* gsrc, unsigned lds_off) {
  unsigned long long ga = (unsigned long long)(size_t)gsrc;
  unsigned ga_lo = (unsigned)__builtin_amdgcn_readfirstlane((int)(ga & 0xffffffffu));
  unsigned ga_hi = (unsigned)__builtin_amdgcn_readfirstlane((int)(ga >> 32));
  u32x4 g0;
  g0[0] = 1u;                                  // count=1 (valid user D#)
  g0[1] = lds_off;                             // lds_addr (bytes)
  g0[2] = ga_lo;                               // global_addr[31:0]
  g0[3] = (ga_hi & 0x01ffffffu) | (2u << 30);  // global_addr[56:32] | type=2
  i32x8 g1;
  g1[0] = (3 << 16);                           // workgroup_mask=0, data_size=8B
  g1[1] = (int)((16384u & 0xffffu) << 16);     // tensor_dim0[15:0]  @bits 63:48
  g1[2] = (int)(16384u >> 16) | (1 << 16);     // tensor_dim0[31:16] | tensor_dim1=1
  g1[3] = (int)(16384u << 16);                 // tile_dim0=16384    @bits 127:112
  g1[4] = 0;                                   // tile_dim1=0, tile_dim2=0
  g1[5] = 16384;                               // tensor_dim0_stride[31:0]
  g1[6] = 0;
  g1[7] = 0;
  i32x4 z4 = {0, 0, 0, 0};
#if defined(__clang_major__) && (__clang_major__ >= 23)
  i32x8 z8 = {0, 0, 0, 0, 0, 0, 0, 0};
  __builtin_amdgcn_tensor_load_to_lds(g0, g1, z4, z4, z8, 0);
#else
  __builtin_amdgcn_tensor_load_to_lds(g0, g1, z4, z4, 0);
#endif
}
#endif // USE_TDM

// ---------------------------------------------------------------------------
// Pack W (fp32 row-major [rows, 2048]) into WMMA B-operand lane layout (fp16).
// Packed flat index: (((ntile*ktiles + kt)*32 + lane)*16 + e)
//   K = kt*32 + (lane/16)*8 + (e%8) + (e/8)*16 ;  N = ntile*16 + (lane%16)
// ---------------------------------------------------------------------------
__global__ void pack_b_kernel(const float* __restrict__ W, _Float16* __restrict__ out,
                              int rowoff, int ktiles) {
  int i = blockIdx.x * 256 + threadIdx.x;
  int total = NTILES * ktiles * 512;
  if (i >= total) return;
  int e     = i & 15;
  int lane  = (i >> 4) & 31;
  int kt    = (i >> 9) % ktiles;
  int ntile = i / (ktiles << 9);
  int k = kt * 32 + (lane >> 4) * 8 + (e & 7) + (e >> 3) * 16;
  int n = (ntile << 4) + (lane & 15);
  out[i] = (_Float16)W[(size_t)(rowoff + k) * GATES + n];
}

// ---------------------------------------------------------------------------
// Embedding gather + fp32 -> fp16 cast.  X layout: [t*16 + b][D_MODEL]
// ---------------------------------------------------------------------------
__global__ void embed_cast_kernel(const int* __restrict__ ids,
                                  const float* __restrict__ table,
                                  _Float16* __restrict__ X) {
  size_t i = (size_t)blockIdx.x * 256 + threadIdx.x;   // one thread per 8 elems
  int    d8  = (int)(i & 127);                         // D_MODEL/8 = 128
  size_t row = i >> 7;                                 // t*16 + b
  int    b   = (int)(row & 15);
  size_t t   = row >> 4;
  int id = ids[(size_t)b * SEQ + t];
  const float4* src = (const float4*)(table + (size_t)id * D_MODEL + d8 * 8);
  float4 x0 = src[0], x1 = src[1];
  union { uint4 u; _Float16 h[8]; } o;
  o.h[0] = (_Float16)x0.x; o.h[1] = (_Float16)x0.y;
  o.h[2] = (_Float16)x0.z; o.h[3] = (_Float16)x0.w;
  o.h[4] = (_Float16)x1.x; o.h[5] = (_Float16)x1.y;
  o.h[6] = (_Float16)x1.z; o.h[7] = (_Float16)x1.w;
  *(uint4*)(X + row * D_MODEL + d8 * 8) = o.u;
}

// ---------------------------------------------------------------------------
// Input-projection GEMM:  gates[t][ntile] = A[t*16.., :K] @ WxPacked + bias
// KTILES compile-time (16 or 32): fully unrolled, 4 independent accumulator
// chains so adjacent WMMAs never depend on each other.
// ---------------------------------------------------------------------------
template <int KTILES>
__global__ __launch_bounds__(256)
void xgemm_kernel(const _Float16* __restrict__ A,
                  const _Float16* __restrict__ Bp,
                  const float* __restrict__ bias,
                  float* __restrict__ gates) {
  constexpr int K = KTILES * 32;
  const int wave = threadIdx.x >> 5;
  const int l    = threadIdx.x & 31;
  const int lm   = l & 15;
  const int lh   = l >> 4;
  const int tile = blockIdx.x * 8 + wave;
  const int t     = tile >> 7;       // M-tile == timestep (B=16)
  const int ntile = tile & 127;

  const _Float16* arow  = A + (size_t)(t * 16 + lm) * K + lh * 8;
  const _Float16* bbase = Bp + (size_t)ntile * KTILES * 512 + (size_t)l * 16;

  v8f acc[4];
#pragma unroll
  for (int j = 0; j < 4; ++j) acc[j] = v8f{0.f, 0.f, 0.f, 0.f, 0.f, 0.f, 0.f, 0.f};

#pragma unroll
  for (int kt = 0; kt < KTILES; ++kt) {
    H16 a, b;
    const _Float16* ap = arow + kt * 32;
    const _Float16* bp = bbase + (size_t)kt * 512;
    a.u[0] = *(const uint4*)(ap);
    a.u[1] = *(const uint4*)(ap + 16);
    b.u[0] = *(const uint4*)(bp);
    b.u[1] = *(const uint4*)(bp + 8);
    if (kt + 2 < KTILES)
      __builtin_prefetch(bp + 1024, 0, 1);   // global_prefetch_b8, k-tile + 2
    acc[kt & 3] = wmma_f16(a.v, b.v, acc[kt & 3]);
  }

  const float bn = bias[ntile * 16 + lm];
#pragma unroll
  for (int v = 0; v < 8; ++v)
    acc[0][v] = (acc[0][v] + acc[1][v]) + (acc[2][v] + acc[3][v]) + bn;

  F8 r; r.f = acc[0];
  uint4* gp = (uint4*)(gates + (((size_t)t * NTILES + ntile) * 32 + l) * 8);
  gp[0] = r.u[0];
  gp[1] = r.u[1];
}

// ---------------------------------------------------------------------------
// Recurrent scan. grid.x = 2 (block 0 = forward, block 1 = reverse).
// 1024 threads = 32 waves; wave w owns hidden cols [16w, 16w+16).
// c-state resident in accumulator VGPRs; h DOUBLE-BUFFERED in LDS so each
// step needs only one barrier and A-tiles are re-read right at the WMMA
// (low register pressure -> no scratch spills).  Wh streams from L2 every
// step: an opaque zero OFFSET (not an opaque pointer, which would break
// address-space inference and turn the loads into flat_load) is redefined
// by empty asm each iteration so the 64 loop-invariant tile loads cannot be
// hoisted out of the time loop into (spilled) registers.
// Dynamic LDS (288 KB of 320 KB):
//   [0,16K)      h buffer 0        [16K,32K)  h buffer 1
//   [32K,160K)   gate buffer 0     [160K,288K) gate buffer 1   (TDM targets)
// ---------------------------------------------------------------------------
__global__ __launch_bounds__(1024)
void lstm_scan_kernel(const float* __restrict__ gatesF, const float* __restrict__ gatesB,
                      const _Float16* __restrict__ whpF, const _Float16* __restrict__ whpB,
                      _Float16* __restrict__ h16F, _Float16* __restrict__ h16B,
                      float* __restrict__ out32, int colF, int colB) {
  const int dir = blockIdx.x;                 // 0 = fwd, 1 = bwd (reverse scan)
  const float*    gates = dir ? gatesB : gatesF;
  const _Float16* whp   = dir ? whpB   : whpF;
  _Float16*       h16   = dir ? h16B   : h16F;
  const int colofs      = dir ? colB   : colF;

  const int tid = threadIdx.x;
  const int w   = tid >> 5;                   // wave id: hidden tile
  const int l   = tid & 31;
  const int lm  = l & 15;
  const int lh  = l >> 4;

  extern __shared__ char smem[];

  for (int i = tid; i < (int)(HBUF_BYTES / 16); i += 1024)   // zero h buffer 0
    ((uint4*)smem)[i] = uint4{0u, 0u, 0u, 0u};

#if USE_TDM
  // preload gate block for the first timestep into gate buffer 0
  if (tid < 32) {
    const int t0 = dir ? (SEQ - 1) : 0;
    tdm_load_gates(gates + (size_t)t0 * NTILES * 256, GBUF_OFF);
    __builtin_amdgcn_s_wait_tensorcnt(0);
  }
#endif
  __syncthreads();

  v8f c = {0.f, 0.f, 0.f, 0.f, 0.f, 0.f, 0.f, 0.f};

  // wave-local base of packed Wh: ntile = g*32 + w; per-gate stride 32 ntiles
  const _Float16* whw = whp + (size_t)w * 16 * 512 + (size_t)l * 16;
  constexpr size_t WH_GSTRIDE = (size_t)32 * 16 * 512;   // halfs per gate group

  for (int s = 0; s < SEQ; ++s) {
    const int t = dir ? (SEQ - 1 - s) : s;

#if USE_TDM
    // kick off DMA of next step's gates into the alternate buffer
    if ((tid < 32) && (s + 1 < SEQ)) {
      const int tn = dir ? (SEQ - 2 - s) : (s + 1);
      tdm_load_gates(gates + (size_t)tn * NTILES * 256,
                     GBUF_OFF + ((unsigned)((s + 1) & 1)) * GSTEP_BYTES);
    }
#endif
    // h double buffers selected by runtime pointer arithmetic on smem
    const unsigned rsel = (unsigned)(s & 1);
    const _Float16* hr = (const _Float16*)(smem + rsel * HBUF_BYTES);        // h(t-1)
    _Float16*       hw = (_Float16*)(smem + (rsel ^ 1u) * HBUF_BYTES);       // h(t)

    // opaque zero offset, redefined per iteration: the Wh loads become
    // loop-variant (no LICM/spill) but keep their global address space.
    unsigned long long zofs = 0;
    asm volatile("" : "+s"(zofs));
    const _Float16* whs = whw + zofs;

    // ---- x-gate pre-activations (fp32, tile-lane order) ----
    v8f acc[4];
#pragma unroll
    for (int g = 0; g < 4; ++g) {
      const int ntile = g * 32 + w;
#if USE_TDM
      const float* gcur = (const float*)(smem + GBUF_OFF + rsel * GSTEP_BYTES);
      const uint4* gp = (const uint4*)(gcur + ((size_t)ntile * 32 + l) * 8);
#else
      const uint4* gp =
          (const uint4*)(gates + (((size_t)t * NTILES + ntile) * 32 + l) * 8);
#endif
      F8 x; x.u[0] = gp[0]; x.u[1] = gp[1];
      acc[g] = x.f;
    }

    // ---- h @ Wh : k-outer, gate-inner -> 4 independent WMMA chains ----
#pragma unroll
    for (int kt = 0; kt < 16; ++kt) {
      H16 a;
      const _Float16* hp = hr + lm * HIDDEN + kt * 32 + lh * 8;
      a.u[0] = *(const uint4*)(hp);
      a.u[1] = *(const uint4*)(hp + 16);
      const _Float16* bk = whs + (size_t)kt * 512;
#pragma unroll
      for (int g = 0; g < 4; ++g) {
        H16 b;
        const _Float16* bp = bk + (size_t)g * WH_GSTRIDE;
        b.u[0] = *(const uint4*)(bp);
        b.u[1] = *(const uint4*)(bp + 8);
        acc[g] = wmma_f16(a.v, b.v, acc[g]);
      }
    }

    // ---- elementwise LSTM cell (c stays in VGPRs) ----
#pragma unroll
    for (int v = 0; v < 8; ++v) {
      const float gi = sigmoidf_(acc[0][v]);
      const float gg = tanhf_(acc[1][v]);
      const float gf = sigmoidf_(acc[2][v] + 1.0f);   // haiku forget bias +1
      const float go = sigmoidf_(acc[3][v]);
      const float cn = gf * c[v] + gi * gg;
      c[v] = cn;
      const float hv = go * tanhf_(cn);

      const int m   = v + lh * 8;        // batch row
      const int hid = w * 16 + lm;       // hidden column
      const _Float16 hh = (_Float16)hv;
      hw[m * HIDDEN + hid] = hh;
      if (h16)   // layer-0: fp16 h stream feeding layer-1 xgemm
        h16[((size_t)t * 16 + m) * HIDDEN + hid] = hh;
      if (out32) // layer-1: fp32 straight into d_out [B,T,1024]
        out32[(size_t)m * ((size_t)SEQ * D_MODEL) + (size_t)t * D_MODEL +
              colofs + hid] = hv;
    }

#if USE_TDM
    if (tid < 32) __builtin_amdgcn_s_wait_tensorcnt(0);  // next gates landed
#endif
    __syncthreads();   // h(t) + next gate buffer visible to all waves
  }
}

// ---------------------------------------------------------------------------
// Host-side orchestration (graph-capture safe: launches only, on `stream`)
// ---------------------------------------------------------------------------
extern "C" void kernel_launch(void* const* d_in, const int* in_sizes, int n_in,
                              void* d_out, int out_size, void* d_ws, size_t ws_size,
                              hipStream_t stream) {
  (void)in_sizes; (void)n_in; (void)out_size; (void)ws_size;

  const int*   ids    = (const int*)  d_in[0];
  const float* table  = (const float*)d_in[1];
  const float* fW0    = (const float*)d_in[2];
  const float* fb0    = (const float*)d_in[3];
  const float* fW1    = (const float*)d_in[4];
  const float* fb1    = (const float*)d_in[5];
  const float* bW0    = (const float*)d_in[6];
  const float* bb0    = (const float*)d_in[7];
  const float* bW1    = (const float*)d_in[8];
  const float* bb1    = (const float*)d_in[9];
  float* out = (float*)d_out;

  // ---- workspace carve-up ----
  char*  base = (char*)d_ws;
  size_t off  = 0;
  auto alloc = [&](size_t bytes) -> char* {
    char* p = base + off;
    off += (bytes + 255) & ~(size_t)255;
    return p;
  };
  _Float16* X      = (_Float16*)alloc((size_t)SEQ * BATCH * D_MODEL * 2); // 64 MB
  _Float16* h0f    = (_Float16*)alloc((size_t)SEQ * BATCH * HIDDEN * 2);  // 32 MB
  _Float16* h0b    = (_Float16*)alloc((size_t)SEQ * BATCH * HIDDEN * 2);  // 32 MB
  _Float16* pWxL0f = (_Float16*)alloc((size_t)NTILES * 32 * 512 * 2);     //  4 MB
  _Float16* pWxL0b = (_Float16*)alloc((size_t)NTILES * 32 * 512 * 2);
  _Float16* pWxL1f = (_Float16*)alloc((size_t)NTILES * 16 * 512 * 2);     //  2 MB
  _Float16* pWxL1b = (_Float16*)alloc((size_t)NTILES * 16 * 512 * 2);
  _Float16* pWhL0f = (_Float16*)alloc((size_t)NTILES * 16 * 512 * 2);
  _Float16* pWhL0b = (_Float16*)alloc((size_t)NTILES * 16 * 512 * 2);
  _Float16* pWhL1f = (_Float16*)alloc((size_t)NTILES * 16 * 512 * 2);
  _Float16* pWhL1b = (_Float16*)alloc((size_t)NTILES * 16 * 512 * 2);
  float* gatesF    = (float*)alloc((size_t)SEQ * NTILES * 256 * 4);       // 256 MB
  float* gatesB    = (float*)alloc((size_t)SEQ * NTILES * 256 * 4);       // 256 MB

  // ---- 1. pack weights into WMMA B-operand layout ----
  const int pk32 = (NTILES * 32 * 512 + 255) / 256;
  const int pk16 = (NTILES * 16 * 512 + 255) / 256;
  pack_b_kernel<<<pk32, 256, 0, stream>>>(fW0, pWxL0f, 0,       32);
  pack_b_kernel<<<pk32, 256, 0, stream>>>(bW0, pWxL0b, 0,       32);
  pack_b_kernel<<<pk16, 256, 0, stream>>>(fW0, pWhL0f, D_MODEL, 16);
  pack_b_kernel<<<pk16, 256, 0, stream>>>(bW0, pWhL0b, D_MODEL, 16);
  pack_b_kernel<<<pk16, 256, 0, stream>>>(fW1, pWxL1f, 0,       16);
  pack_b_kernel<<<pk16, 256, 0, stream>>>(bW1, pWxL1b, 0,       16);
  pack_b_kernel<<<pk16, 256, 0, stream>>>(fW1, pWhL1f, HIDDEN,  16);
  pack_b_kernel<<<pk16, 256, 0, stream>>>(bW1, pWhL1b, HIDDEN,  16);

  // ---- 2. embedding gather + cast ----
  const size_t embThreads = (size_t)SEQ * BATCH * (D_MODEL / 8);
  embed_cast_kernel<<<(int)(embThreads / 256), 256, 0, stream>>>(ids, table, X);

  const int gemmBlocks = SEQ * NTILES / 8;   // 32768

  // ---- 3. layer-0 input projections (K = 1024) ----
  xgemm_kernel<32><<<gemmBlocks, 256, 0, stream>>>(X, pWxL0f, fb0, gatesF);
  xgemm_kernel<32><<<gemmBlocks, 256, 0, stream>>>(X, pWxL0b, bb0, gatesB);

  // ---- 4. layer-0 recurrent scans (fwd + bwd concurrently) ----
  lstm_scan_kernel<<<2, 1024, SCAN_LDS, stream>>>(gatesF, gatesB, pWhL0f, pWhL0b,
                                                  h0f, h0b, nullptr, 0, 0);

  // ---- 5. layer-1 input projections (K = 512) ----
  xgemm_kernel<16><<<gemmBlocks, 256, 0, stream>>>(h0f, pWxL1f, fb1, gatesF);
  xgemm_kernel<16><<<gemmBlocks, 256, 0, stream>>>(h0b, pWxL1b, bb1, gatesB);

  // ---- 6. layer-1 recurrent scans -> d_out [B,T,1024] ----
  lstm_scan_kernel<<<2, 1024, SCAN_LDS, stream>>>(gatesF, gatesB, pWhL1f, pWhL1b,
                                                  nullptr, nullptr, out, 0, HIDDEN);
}
